// global_feature_information_60885456388615
// MI455X (gfx1250) — matrix operand order repaired and checked by reading, not compiled, not run on hardware
//
#include <hip/hip_runtime.h>
#include <cstdint>
#include <cstddef>

// ---------------------------------------------------------------------------
// CDNA5 (gfx1250) Swin-T forward. wave32. WMMA bf16 -> f32 accumulate.
// Operand buffers are bf16 (producers write bf16); residual stream is f32.
// GEMM tiles stream to LDS via GLOBAL_LOAD_ASYNC_TO_LDS_B128 (double-buffered).
// ---------------------------------------------------------------------------

typedef __bf16 bf16_t;
typedef bf16_t v16bf __attribute__((ext_vector_type(16)));
typedef bf16_t v8bf  __attribute__((ext_vector_type(8)));
typedef float  v8f   __attribute__((ext_vector_type(8)));

#if defined(__has_builtin)
#if __has_builtin(__builtin_amdgcn_global_load_async_to_lds_b128)
#define HAVE_ASYNC_LDS 1
#endif
#endif
#ifndef HAVE_ASYNC_LDS
#define HAVE_ASYNC_LDS 0
#endif

#if HAVE_ASYNC_LDS
// Builtin parameter type (from hipcc diagnostic): pointer to 4 x int vector.
typedef int v4i_t __attribute__((__vector_size__(4 * sizeof(int))));
typedef __attribute__((address_space(1))) v4i_t as1_v4i;
typedef __attribute__((address_space(3))) v4i_t as3_v4i;

__device__ __forceinline__ void async_cp16(const bf16_t* g, bf16_t* l) {
  __builtin_amdgcn_global_load_async_to_lds_b128((as1_v4i*)(g), (as3_v4i*)(l),
                                                 0, 0);
}
__device__ __forceinline__ void wait_async0() {
#if __has_builtin(__builtin_amdgcn_s_wait_asynccnt)
  __builtin_amdgcn_s_wait_asynccnt(0);
#else
  asm volatile("s_wait_asynccnt 0x0" ::: "memory");
#endif
}
#endif

// ---- WMMA fragment loaders (per CDNA5 ISA 7.12.2 VGPR layouts) -------------
// A (16x32 bf16, MxK): lanes 0-15 rows M=lane; K chunks {0..7,16..23} (+8 for
// lanes 16-31). Two contiguous 16B chunks -> two ds_load_b128.
__device__ __forceinline__ v16bf frag_a_ld(const bf16_t* base, int stride,
                                           int lane) {
  const int r = lane & 15;
  const int hf = (lane >> 4) * 8;
  const bf16_t* p = base + r * stride + hf;
  v8bf lo = *(const v8bf*)(p);        // K = hf+0..7
  v8bf hi = *(const v8bf*)(p + 16);   // K = hf+16..23
  return __builtin_shufflevector(lo, hi, 0, 1, 2, 3, 4, 5, 6, 7, 8, 9, 10, 11,
                                 12, 13, 14, 15);
}

// B (32x16 bf16, KxN) stored transposed (N x K row-major) in LDS:
// lane n holds column N=lane&15; K chunks {0..7,16..23} (+8 for lanes 16-31).
__device__ __forceinline__ v16bf frag_b_ld(const bf16_t* base, int stride,
                                           int lane) {
  const int n = lane & 15;
  const int hf = (lane >> 4) * 8;
  const bf16_t* p = base + n * stride + hf;
  v8bf lo = *(const v8bf*)(p);
  v8bf hi = *(const v8bf*)(p + 16);
  return __builtin_shufflevector(lo, hi, 0, 1, 2, 3, 4, 5, 6, 7, 8, 9, 10, 11,
                                 12, 13, 14, 15);
}

__device__ __forceinline__ v8f zero_v8f() {
  v8f z = {0.f, 0.f, 0.f, 0.f, 0.f, 0.f, 0.f, 0.f};
  return z;
}

__device__ __forceinline__ void pack16(const float4 q[4], v8bf& lo, v8bf& hi) {
  lo[0] = (bf16_t)q[0].x; lo[1] = (bf16_t)q[0].y;
  lo[2] = (bf16_t)q[0].z; lo[3] = (bf16_t)q[0].w;
  lo[4] = (bf16_t)q[1].x; lo[5] = (bf16_t)q[1].y;
  lo[6] = (bf16_t)q[1].z; lo[7] = (bf16_t)q[1].w;
  hi[0] = (bf16_t)q[2].x; hi[1] = (bf16_t)q[2].y;
  hi[2] = (bf16_t)q[2].z; hi[3] = (bf16_t)q[2].w;
  hi[4] = (bf16_t)q[3].x; hi[5] = (bf16_t)q[3].y;
  hi[6] = (bf16_t)q[3].z; hi[7] = (bf16_t)q[3].w;
}

// ---------------------------------------------------------------------------
// Generic GEMM: Y[M][N] = epi( A[M][K] * W[N][K]^T + bias ) (+ res)
// A, W are bf16; Y is f32 or bf16 (OutT). epi: 0 none, 1 +bias,
// 2 +bias+GELU, 3 +bias+residual. Block = 128 threads (4 waves), 64x64 tile,
// each wave 32x32 (2x2 WMMA). K is a multiple of 32; OOB rows clamped (their
// products only feed never-stored outputs) -> branch-free tile fills.
// ---------------------------------------------------------------------------
template <typename OutT>
__global__ __launch_bounds__(128) void gemm_wmma_bf16(
    const bf16_t* __restrict__ A, const bf16_t* __restrict__ Wt,
    const float* __restrict__ bias, const float* __restrict__ res,
    OutT* __restrict__ Y, int M, int N, int K, int epi) {
#if HAVE_ASYNC_LDS
  __shared__ __align__(16) bf16_t As[2][64][40];
  __shared__ __align__(16) bf16_t Bs[2][64][40];
#else
  __shared__ __align__(16) bf16_t As[1][64][40];
  __shared__ __align__(16) bf16_t Bs[1][64][40];
#endif
  const int bm = blockIdx.x * 64;
  const int bn = blockIdx.y * 64;
  const int tid = threadIdx.x;
  const int lane = tid & 31;
  const int wv = tid >> 5;
  const int wm = (wv >> 1) * 32;
  const int wn = (wv & 1) * 32;

  v8f acc[2][2];
#pragma unroll
  for (int i = 0; i < 2; ++i)
#pragma unroll
    for (int j = 0; j < 2; ++j) acc[i][j] = zero_v8f();

  const int lrow = tid >> 1;        // 0..63
  const int lseg = (tid & 1) * 16;  // 0 or 16
  const int gmc = min(bm + lrow, M - 1);
  const int gnc = min(bn + lrow, N - 1);
  const bf16_t* Abase = A + (size_t)gmc * K + lseg;
  const bf16_t* Bbase = Wt + (size_t)gnc * K + lseg;

#if HAVE_ASYNC_LDS
  // stream tile 0 into buffer 0
  async_cp16(Abase, &As[0][lrow][lseg]);
  async_cp16(Abase + 8, &As[0][lrow][lseg + 8]);
  async_cp16(Bbase, &Bs[0][lrow][lseg]);
  async_cp16(Bbase + 8, &Bs[0][lrow][lseg + 8]);
  int pb = 0;
  for (int k0 = 0; k0 < K; k0 += 32) {
    wait_async0();      // this wave's tile (and any previous issue) landed
    __syncthreads();    // all waves' tiles landed; prev buffer free to refill
    if (k0 + 32 < K) {
      const bf16_t* an = Abase + k0 + 32;
      const bf16_t* bn2 = Bbase + k0 + 32;
      async_cp16(an, &As[pb ^ 1][lrow][lseg]);
      async_cp16(an + 8, &As[pb ^ 1][lrow][lseg + 8]);
      async_cp16(bn2, &Bs[pb ^ 1][lrow][lseg]);
      async_cp16(bn2 + 8, &Bs[pb ^ 1][lrow][lseg + 8]);
    }
#pragma unroll
    for (int tm = 0; tm < 2; ++tm) {
      v16bf af = frag_a_ld(&As[pb][wm + tm * 16][0], 40, lane);
#pragma unroll
      for (int tn = 0; tn < 2; ++tn) {
        v16bf bfr = frag_b_ld(&Bs[pb][wn + tn * 16][0], 40, lane);
        acc[tm][tn] = __builtin_amdgcn_wmma_f32_16x16x32_bf16(
            false, af, false, bfr, (short)0, acc[tm][tn], false, false);
      }
    }
    pb ^= 1;
  }
#else
  v8bf arl, arh, brl, brh;
  arl = *(const v8bf*)(Abase);
  arh = *(const v8bf*)(Abase + 8);
  brl = *(const v8bf*)(Bbase);
  brh = *(const v8bf*)(Bbase + 8);
  for (int k0 = 0; k0 < K; k0 += 32) {
    *(v8bf*)&As[0][lrow][lseg] = arl;
    *(v8bf*)&As[0][lrow][lseg + 8] = arh;
    *(v8bf*)&Bs[0][lrow][lseg] = brl;
    *(v8bf*)&Bs[0][lrow][lseg + 8] = brh;
    __syncthreads();
    if (k0 + 32 < K) {
      arl = *(const v8bf*)(Abase + k0 + 32);
      arh = *(const v8bf*)(Abase + k0 + 40);
      brl = *(const v8bf*)(Bbase + k0 + 32);
      brh = *(const v8bf*)(Bbase + k0 + 40);
      if (k0 + 64 < K) {
        __builtin_prefetch(Abase + k0 + 64, 0, 3);
        __builtin_prefetch(Bbase + k0 + 64, 0, 3);
      }
    }
#pragma unroll
    for (int tm = 0; tm < 2; ++tm) {
      v16bf af = frag_a_ld(&As[0][wm + tm * 16][0], 40, lane);
#pragma unroll
      for (int tn = 0; tn < 2; ++tn) {
        v16bf bfr = frag_b_ld(&Bs[0][wn + tn * 16][0], 40, lane);
        acc[tm][tn] = __builtin_amdgcn_wmma_f32_16x16x32_bf16(
            false, af, false, bfr, (short)0, acc[tm][tn], false, false);
      }
    }
    __syncthreads();
  }
#endif

  const int hf = lane >> 4;
  const int ln = lane & 15;
#pragma unroll
  for (int tm = 0; tm < 2; ++tm) {
#pragma unroll
    for (int tn = 0; tn < 2; ++tn) {
      const int gn = bn + wn + tn * 16 + ln;
#pragma unroll
      for (int i = 0; i < 8; ++i) {
        const int gm = bm + wm + tm * 16 + i + hf * 8;
        if (gm < M && gn < N) {
          float v = acc[tm][tn][i];
          if (epi >= 1 && bias) v += bias[gn];
          if (epi == 2) v = 0.5f * v * (1.0f + erff(v * 0.70710678118654752f));
          if (epi == 3 && res) v += res[(size_t)gm * N + gn];
          Y[(size_t)gm * N + gn] = (OutT)v;
        }
      }
    }
  }
}

// ---------------------------------------------------------------------------
// Fused window attention: one block = one (window, head). 49 tokens -> pad 64.
// head_dim == 32 in every Swin-T stage -> single K-step for Q*K^T.
// qkv input is f32 (GEMM epilogue output); attention output is bf16.
// Rel-bias slice for this head is preloaded into LDS; S epilogue branchless.
// ---------------------------------------------------------------------------
__global__ __launch_bounds__(128) void attn_win(
    const float* __restrict__ qkv, const float* __restrict__ rel,
    bf16_t* __restrict__ out, int heads, int C, int winPerImg, int nw, int H,
    int W, int shift) {
  const int win = blockIdx.x / heads;
  const int head = blockIdx.x % heads;

  __shared__ __align__(16) bf16_t Qs[64][40];
  __shared__ __align__(16) bf16_t Ks[64][40];
  __shared__ __align__(16) bf16_t Vt[32][72];  // Vt[n][k] = V[k][n]
  __shared__ float Ss[64][65];
  __shared__ __align__(16) bf16_t Ps[64][72];
  __shared__ float relS[169];                  // rel bias slice for this head

  const int tid = threadIdx.x;
  const int lane = tid & 31;
  const int wv = tid >> 5;
  const float scale = 0.17677669529663687f;  // 32^-0.5

  // preload rel table slice (13x13)
  for (int i = tid; i < 169; i += 128) relS[i] = rel[i * heads + head];

  {
    const int row = tid >> 1;
    const int seg = (tid & 1) * 16;
    if (row < 49) {
      const float* bq = qkv + (size_t)(win * 49 + row) * (3 * C) + head * 32;
      float4 q[4];
      const float4* pq = (const float4*)(bq + seg);
      const float4* pk = (const float4*)(bq + C + seg);
      const float4* pv = (const float4*)(bq + 2 * C + seg);
      v8bf lo, hi;
#pragma unroll
      for (int i = 0; i < 4; ++i) q[i] = pq[i];
#pragma unroll
      for (int i = 0; i < 4; ++i) {
        q[i].x *= scale; q[i].y *= scale; q[i].z *= scale; q[i].w *= scale;
      }
      pack16(q, lo, hi);
      *(v8bf*)&Qs[row][seg] = lo;
      *(v8bf*)&Qs[row][seg + 8] = hi;
#pragma unroll
      for (int i = 0; i < 4; ++i) q[i] = pk[i];
      pack16(q, lo, hi);
      *(v8bf*)&Ks[row][seg] = lo;
      *(v8bf*)&Ks[row][seg + 8] = hi;
#pragma unroll
      for (int i = 0; i < 4; ++i) q[i] = pv[i];
      const float* vf = (const float*)q;
#pragma unroll
      for (int e = 0; e < 16; ++e) Vt[seg + e][row] = (bf16_t)vf[e];
    } else {
      v8bf z;
#pragma unroll
      for (int e = 0; e < 8; ++e) z[e] = (bf16_t)0.f;
      *(v8bf*)&Qs[row][seg] = z;
      *(v8bf*)&Qs[row][seg + 8] = z;
      *(v8bf*)&Ks[row][seg] = z;
      *(v8bf*)&Ks[row][seg + 8] = z;
#pragma unroll
      for (int e = 0; e < 16; ++e) Vt[seg + e][row] = (bf16_t)0.f;
    }
  }
  __syncthreads();

  // S = Q * K^T  (wave wv owns rows [wv*16, wv*16+16) )
  const int m0 = wv * 16;
  const int wl = win % winPerImg;
  const int wh = wl / nw, wwi = wl % nw;
  {
    v16bf af = frag_a_ld(&Qs[m0][0], 40, lane);
    // hoist row-side terms (8 rows owned by this lane, shared by all 4 tiles)
    int riA[8], ciA[8], giA[8];
    bool rvA[8];
#pragma unroll
    for (int i = 0; i < 8; ++i) {
      const int row = m0 + i + (lane >> 4) * 8;
      rvA[i] = row < 49;
      const int rc = min(row, 48);
      riA[i] = rc / 7;
      ciA[i] = rc % 7;
      const int hi = wh * 7 + riA[i], wi = wwi * 7 + ciA[i];
      giA[i] = (hi < H - 7 ? 0 : (hi < H - shift ? 1 : 2)) * 3 +
               (wi < W - 7 ? 0 : (wi < W - shift ? 1 : 2));
    }
#pragma unroll
    for (int tn = 0; tn < 4; ++tn) {
      v16bf bfr = frag_b_ld(&Ks[tn * 16][0], 40, lane);
      v8f acc = zero_v8f();
      acc = __builtin_amdgcn_wmma_f32_16x16x32_bf16(false, af, false, bfr,
                                                    (short)0, acc, false, false);
      const int col = tn * 16 + (lane & 15);
      const bool cv = col < 49;
      const int cc2 = min(col, 48);
      const int rj = cc2 / 7, cj = cc2 % 7;
      const int hj = wh * 7 + rj, wj = wwi * 7 + cj;
      const int gj = (hj < H - 7 ? 0 : (hj < H - shift ? 1 : 2)) * 3 +
                     (wj < W - 7 ? 0 : (wj < W - shift ? 1 : 2));
#pragma unroll
      for (int i = 0; i < 8; ++i) {
        const int row = m0 + i + (lane >> 4) * 8;
        const int ridx = (riA[i] - rj + 6) * 13 + (ciA[i] - cj + 6);
        float v = acc[i] + relS[ridx];
        if (shift && giA[i] != gj) v -= 100.f;
        Ss[row][col] = (rvA[i] && cv) ? v : 0.f;
      }
    }
  }
  __syncthreads();

  // row softmax over the 49 valid columns; P padded with zeros.
  if (tid < 49) {
    float mx = -1e30f;
    for (int c = 0; c < 49; ++c) mx = fmaxf(mx, Ss[tid][c]);
    float sum = 0.f;
    for (int c = 0; c < 49; ++c) {
      const float e = __expf(Ss[tid][c] - mx);
      sum += e;
      Ss[tid][c] = e;
    }
    const float inv = 1.f / sum;
    for (int c = 0; c < 49; ++c) Ps[tid][c] = (bf16_t)(Ss[tid][c] * inv);
    for (int c = 49; c < 64; ++c) Ps[tid][c] = (bf16_t)0.f;
  } else if (tid < 64) {
    for (int c = 0; c < 64; ++c) Ps[tid][c] = (bf16_t)0.f;
  }
  __syncthreads();

  // O = P * V  (M=64, N=32, K=64 as two 32-steps)
  v8f ov[2];
  ov[0] = zero_v8f();
  ov[1] = zero_v8f();
#pragma unroll
  for (int ks = 0; ks < 2; ++ks) {
    v16bf pa = frag_a_ld(&Ps[m0][ks * 32], 72, lane);
#pragma unroll
    for (int tn = 0; tn < 2; ++tn) {
      v16bf vb = frag_b_ld(&Vt[tn * 16][ks * 32], 72, lane);
      ov[tn] = __builtin_amdgcn_wmma_f32_16x16x32_bf16(false, pa, false, vb,
                                                       (short)0, ov[tn], false,
                                                       false);
    }
  }
#pragma unroll
  for (int tn = 0; tn < 2; ++tn) {
    const int col = tn * 16 + (lane & 15);
#pragma unroll
    for (int i = 0; i < 8; ++i) {
      const int row = m0 + i + (lane >> 4) * 8;
      if (row < 49)
        out[(size_t)(win * 49 + row) * C + head * 32 + col] =
            (bf16_t)ov[tn][i];
    }
  }
}

// ---------------------------------------------------------------------------
// Elementwise / data-movement kernels
// ---------------------------------------------------------------------------
__global__ __launch_bounds__(256) void patch_embed_k(
    const float* __restrict__ x, const float* __restrict__ w,
    const float* __restrict__ bias, float* __restrict__ y, int B) {
  const int t = blockIdx.x * blockDim.x + threadIdx.x;
  const int total = B * 56 * 56 * 96;
  if (t >= total) return;
  const int oc = t % 96;
  int r = t / 96;
  const int ow = r % 56;
  r /= 56;
  const int oh = r % 56;
  const int b = r / 56;
  float acc = bias[oc];
#pragma unroll
  for (int ic = 0; ic < 3; ++ic)
#pragma unroll
    for (int kh = 0; kh < 4; ++kh)
#pragma unroll
      for (int kw = 0; kw < 4; ++kw)
        acc += x[(((size_t)b * 3 + ic) * 224 + oh * 4 + kh) * 224 + ow * 4 + kw] *
               w[((oc * 3 + ic) * 4 + kh) * 4 + kw];
  y[t] = acc;
}

// one wave per row; OutT selects f32 (residual stream) or bf16 (GEMM operand)
template <typename OutT>
__global__ __launch_bounds__(256) void layernorm_k(
    const float* __restrict__ x, const float* __restrict__ g,
    const float* __restrict__ b, OutT* __restrict__ y, int R, int C) {
  const int row = blockIdx.x * 8 + (threadIdx.x >> 5);
  const int lane = threadIdx.x & 31;
  if (row >= R) return;
  const float* xr = x + (size_t)row * C;
  float s = 0.f, ss = 0.f;
  for (int c = lane; c < C; c += 32) {
    const float v = xr[c];
    s += v;
    ss += v * v;
  }
#pragma unroll
  for (int o = 16; o > 0; o >>= 1) {
    s += __shfl_xor(s, o, 32);
    ss += __shfl_xor(ss, o, 32);
  }
  const float mu = s / C;
  const float var = ss / C - mu * mu;
  const float inv = rsqrtf(var + 1e-5f);
  OutT* yr = y + (size_t)row * C;
  for (int c = lane; c < C; c += 32)
    yr[c] = (OutT)((xr[c] - mu) * inv * g[c] + b[c]);
}

__global__ __launch_bounds__(256) void f32_to_bf16_k(const float* __restrict__ x,
                                                     bf16_t* __restrict__ y,
                                                     int n) {
  const int t = blockIdx.x * blockDim.x + threadIdx.x;
  const int n2 = n >> 1;
  if (t < n2) {
    const float2 v = ((const float2*)x)[t];
    y[2 * t] = (bf16_t)v.x;
    y[2 * t + 1] = (bf16_t)v.y;
  }
}

// bf16 window partition/copy; C even -> move u32 pairs
__global__ __launch_bounds__(256) void win_partition_k(
    const uint32_t* __restrict__ x, uint32_t* __restrict__ xw, int B, int H,
    int W, int C2, int shift, int nh, int nw) {
  const int t = blockIdx.x * blockDim.x + threadIdx.x;
  const int total = B * H * W * C2;
  if (t >= total) return;
  const int c = t % C2;
  const int tok = t / C2;
  const int i = tok % 49;
  const int win = tok / 49;
  const int ww = win % nw;
  const int r1 = win / nw;
  const int wh = r1 % nh;
  const int b = r1 / nh;
  const int r = i / 7, cc = i % 7;
  const int h = (wh * 7 + r + shift) % H;
  const int w = (ww * 7 + cc + shift) % W;
  xw[t] = x[(((size_t)b * H + h) * W + w) * C2 + c];
}

__global__ __launch_bounds__(256) void win_scatter_add_k(
    const float* __restrict__ p, float* __restrict__ x, int B, int H, int W,
    int C, int shift, int nh, int nw) {
  const int t = blockIdx.x * blockDim.x + threadIdx.x;
  const int total = B * H * W * C;
  if (t >= total) return;
  const int c = t % C;
  const int tok = t / C;
  const int i = tok % 49;
  const int win = tok / 49;
  const int ww = win % nw;
  const int r1 = win / nw;
  const int wh = r1 % nh;
  const int b = r1 / nh;
  const int r = i / 7, cc = i % 7;
  const int h = (wh * 7 + r + shift) % H;
  const int w = (ww * 7 + cc + shift) % W;
  x[(((size_t)b * H + h) * W + w) * C + c] += p[t];
}

__global__ __launch_bounds__(256) void merge_gather_k(
    const float* __restrict__ x, float* __restrict__ y, int B, int H, int W,
    int C) {
  const int t = blockIdx.x * blockDim.x + threadIdx.x;
  const int H2 = H / 2, W2 = W / 2, C4 = 4 * C;
  const int total = B * H2 * W2 * C4;
  if (t >= total) return;
  const int c4 = t % C4;
  const int tok = t / C4;
  const int w2 = tok % W2;
  const int r = tok / W2;
  const int h2 = r % H2;
  const int b = r / H2;
  const int q = c4 / C, c = c4 % C;
  const int dh = (q == 1 || q == 3) ? 1 : 0;
  const int dw = (q >= 2) ? 1 : 0;
  y[t] = x[(((size_t)b * H + 2 * h2 + dh) * W + 2 * w2 + dw) * C + c];
}

__global__ __launch_bounds__(256) void sigmoid_k(float* __restrict__ x, int n) {
  const int t = blockIdx.x * blockDim.x + threadIdx.x;
  if (t < n) x[t] = 1.f / (1.f + __expf(-x[t]));
}

__global__ __launch_bounds__(256) void nchw_out_k(const float* __restrict__ x,
                                                  float* __restrict__ out,
                                                  int B, int H, int W, int C) {
  const int t = blockIdx.x * blockDim.x + threadIdx.x;
  const int total = B * H * W * C;
  if (t >= total) return;
  const int c = t % C;
  const int tok = t / C;
  const int w = tok % W;
  const int h = (tok / W) % H;
  const int b = tok / (W * H);
  out[(((size_t)b * C + c) * H + h) * W + w] = x[t];
}

// ---------------------------------------------------------------------------
// Host orchestration
// ---------------------------------------------------------------------------
extern "C" void kernel_launch(void* const* d_in, const int* in_sizes, int n_in,
                              void* d_out, int out_size, void* d_ws,
                              size_t ws_size, hipStream_t stream) {
  (void)in_sizes;
  (void)n_in;
  (void)out_size;

  int idx = 0;
  auto nxt = [&]() { return (const float*)d_in[idx++]; };

  const float* x_in = nxt();  // (8,3,224,224)

  struct MergeP { const float *ln_b, *ln_g, *red_w; } mg[3];
  for (int m = 0; m < 3; ++m) {
    mg[m].ln_b = nxt();
    mg[m].ln_g = nxt();
    mg[m].red_w = nxt();
  }
  const float* patch_b = nxt();
  const float* patch_ln_b = nxt();
  const float* patch_ln_g = nxt();
  const float* patch_w = nxt();

  struct BlkP {
    const float *fc1_b, *fc1_w, *fc2_b, *fc2_w, *ln1_b, *ln1_g, *ln2_b, *ln2_g,
        *proj_b, *proj_w, *qkv_b, *qkv_w, *rel;
  } blk[12];
  const int nb[4] = {2, 2, 6, 2};
  int bi = 0;
  for (int s = 0; s < 4; ++s)
    for (int b = 0; b < nb[s]; ++b) {
      BlkP& p = blk[bi++];
      p.fc1_b = nxt();  p.fc1_w = nxt();
      p.fc2_b = nxt();  p.fc2_w = nxt();
      p.ln1_b = nxt();  p.ln1_g = nxt();
      p.ln2_b = nxt();  p.ln2_g = nxt();
      p.proj_b = nxt(); p.proj_w = nxt();
      p.qkv_b = nxt();  p.qkv_w = nxt();
      p.rel = nxt();
    }

  // workspace layout (in float units)
  const size_t CAP = 2408448;   // max tokens*C (also tokens2*4C)
  const size_t BIG = 9633792;   // max tokens*4C (MLP hidden / QKV)
  const size_t WCF = 1179648;   // bf16 weight staging (max 2359296 elts)
  if (ws_size < (5 * CAP + BIG + WCF) * sizeof(float)) return;
  float* act0 = (float*)d_ws;
  float* act1 = act0 + CAP;
  bf16_t* lnb = (bf16_t*)(act1 + CAP);   // bf16, <= CAP elts
  float* xwb = act1 + 2 * CAP;           // f32 CAP; also bf16 window buffer
  bf16_t* attb = (bf16_t*)(xwb + CAP);   // bf16, <= CAP elts
  float* qkvb = xwb + 2 * CAP;           // f32 qkv / bf16 hidden (BIG elts)
  bf16_t* wbf = (bf16_t*)(qkvb + BIG);   // bf16 staged weights

  bf16_t* xwb_bf = (bf16_t*)xwb;
  bf16_t* hid_bf = (bf16_t*)qkvb;

  const int Bn = 8;

  auto stage_w = [&](const float* w, int n) {
    f32_to_bf16_k<<<((n >> 1) + 255) / 256, 256, 0, stream>>>(w, wbf, n);
  };

  // --- patch embed + LN (residual stream stays f32) ---
  {
    const int total = Bn * 56 * 56 * 96;
    patch_embed_k<<<(total + 255) / 256, 256, 0, stream>>>(x_in, patch_w,
                                                           patch_b, act0, Bn);
    const int R = Bn * 56 * 56;
    layernorm_k<float><<<(R + 7) / 8, 256, 0, stream>>>(
        act0, patch_ln_g, patch_ln_b, act0, R, 96);
  }

  float* act = act0;
  float* spare = act1;
  int H = 56, W = 56, C = 96;
  const int heads_arr[4] = {3, 6, 12, 24};
  bi = 0;
  for (int s = 0; s < 4; ++s) {
    const int heads = heads_arr[s];
    for (int b = 0; b < nb[s]; ++b) {
      const BlkP& p = blk[bi++];
      const int shift = (b % 2 == 1 && H > 7) ? 3 : 0;
      const int T = Bn * H * W;
      const int nh = H / 7, nw = W / 7;
      const int nWin = Bn * nh * nw;

      // LN1 -> bf16
      layernorm_k<bf16_t><<<(T + 7) / 8, 256, 0, stream>>>(act, p.ln1_g,
                                                           p.ln1_b, lnb, T, C);
      // window partition (bf16 u32-pair copy)
      {
        const int tot = T * (C / 2);
        win_partition_k<<<(tot + 255) / 256, 256, 0, stream>>>(
            (const uint32_t*)lnb, (uint32_t*)xwb_bf, Bn, H, W, C / 2, shift,
            nh, nw);
      }
      // qkv = xw @ Wqkv^T + b   (f32 out for attention)
      stage_w(p.qkv_w, 3 * C * C);
      {
        dim3 g((T + 63) / 64, (3 * C + 63) / 64);
        gemm_wmma_bf16<float><<<g, 128, 0, stream>>>(
            xwb_bf, wbf, p.qkv_b, nullptr, qkvb, T, 3 * C, C, 1);
      }
      // attention (bf16 out)
      attn_win<<<nWin * heads, 128, 0, stream>>>(qkvb, p.rel, attb, heads, C,
                                                 nh * nw, nw, H, W, shift);
      // proj (f32 out, then scatter+residual)
      stage_w(p.proj_w, C * C);
      {
        dim3 g((T + 63) / 64, (C + 63) / 64);
        gemm_wmma_bf16<float><<<g, 128, 0, stream>>>(
            attb, wbf, p.proj_b, nullptr, xwb, T, C, C, 1);
      }
      {
        const int tot = T * C;
        win_scatter_add_k<<<(tot + 255) / 256, 256, 0, stream>>>(
            xwb, act, Bn, H, W, C, shift, nh, nw);
      }
      // LN2 -> bf16
      layernorm_k<bf16_t><<<(T + 7) / 8, 256, 0, stream>>>(act, p.ln2_g,
                                                           p.ln2_b, lnb, T, C);
      // fc1 + GELU (bf16 hidden out)
      stage_w(p.fc1_w, 4 * C * C);
      {
        dim3 g((T + 63) / 64, (4 * C + 63) / 64);
        gemm_wmma_bf16<bf16_t><<<g, 128, 0, stream>>>(
            lnb, wbf, p.fc1_b, nullptr, hid_bf, T, 4 * C, C, 2);
      }
      // fc2 + residual (in-place into act, f32)
      stage_w(p.fc2_w, C * 4 * C);
      {
        dim3 g((T + 63) / 64, (C + 63) / 64);
        gemm_wmma_bf16<float><<<g, 128, 0, stream>>>(
            hid_bf, wbf, p.fc2_b, act, act, T, C, 4 * C, 3);
      }
    }
    {
      const int tot = Bn * H * W * C;
      sigmoid_k<<<(tot + 255) / 256, 256, 0, stream>>>(act, tot);
    }
    if (s < 3) {
      const int H2 = H / 2, W2 = W / 2, T2 = Bn * H2 * W2;
      {
        const int tot = T2 * 4 * C;
        merge_gather_k<<<(tot + 255) / 256, 256, 0, stream>>>(act, xwb, Bn, H,
                                                              W, C);
      }
      layernorm_k<bf16_t><<<(T2 + 7) / 8, 256, 0, stream>>>(
          xwb, mg[s].ln_g, mg[s].ln_b, lnb, T2, 4 * C);
      stage_w(mg[s].red_w, 2 * C * 4 * C);
      {
        dim3 g((T2 + 63) / 64, (2 * C + 63) / 64);
        gemm_wmma_bf16<float><<<g, 128, 0, stream>>>(
            lnb, wbf, nullptr, nullptr, spare, T2, 2 * C, 4 * C, 0);
      }
      float* t = act;
      act = spare;
      spare = t;
      H = H2;
      W = W2;
      C *= 2;
    }
  }

  // final NHWC -> NCHW
  {
    const int tot = Bn * H * W * C;
    nchw_out_k<<<(tot + 255) / 256, 256, 0, stream>>>(act, (float*)d_out, Bn, H,
                                                      W, C);
  }
}
